// Style2ResidualBlock1DBeta_54966991454474
// MI455X (gfx1250) — matrix operand order repaired and checked by reading, not compiled, not run on hardware
//
#include <hip/hip_runtime.h>

#define B_    16
#define CIN   512
#define T_    2048
#define TP    2050          // padded time (1 zero col each side)
#define COUT2 1024
#define CGLU  512
#define KW    3
#define NSTYLE 128
#define RED   1536          // CIN*KW
#define NT    4             // time tiles per wave (N = 64 per wave)

typedef __attribute__((ext_vector_type(16))) __bf16       v16bf;
typedef __attribute__((ext_vector_type(8)))  float        v8f;
typedef __attribute__((ext_vector_type(4)))  unsigned int uint4v;

union Frag { v16bf bf; uint4v u[2]; };

// ---------------- style modulation: s = c_trg @ (w*SCALE_LIN)^T + bias ----------------
__global__ void style_kernel(const float* __restrict__ c_trg,
                             const float* __restrict__ w_s, const float* __restrict__ b_s,
                             const float* __restrict__ w_b, const float* __restrict__ b_b,
                             float* __restrict__ s_out, float* __restrict__ beta_out) {
  int idx = blockIdx.x * blockDim.x + threadIdx.x;          // B_*CIN threads
  if (idx >= B_ * CIN) return;
  int b = idx >> 9, i = idx & (CIN - 1);
  const float* c  = c_trg + b * NSTYLE;
  const float* ws = w_s + i * NSTYLE;
  const float* wb = w_b + i * NSTYLE;
  float as = 0.f, ab = 0.f;
#pragma unroll 8
  for (int j = 0; j < NSTYLE; ++j) { float cv = c[j]; as += cv * ws[j]; ab += cv * wb[j]; }
  const float SCALE_LIN = 0.08838834764831845f;             // 1/sqrt(128)
  s_out[idx]    = as * SCALE_LIN + b_s[i];
  beta_out[idx] = ab * SCALE_LIN + b_b[i];
}

// ---------------- per-(b,o) weight modulation + demod, write bf16 [b][o][k*512+i] ----------------
__global__ __launch_bounds__(256) void modw_kernel(const float* __restrict__ weight,  // [COUT2][CIN][KW]
                            const float* __restrict__ s, const float* __restrict__ beta,
                            __bf16* __restrict__ wmod) {                              // [B_][COUT2][RED]
  int bo = blockIdx.x;                 // b*1024 + o
  int b  = bo >> 10;
  int o  = bo & 1023;
  int tid = threadIdx.x;
  const float SCALE_CONV = 0.014731391274719739f;           // 1/sqrt(512*9)
  float vals[6];
  float sum = 0.f, sq = 0.f;
#pragma unroll
  for (int jj = 0; jj < 6; ++jj) {
    int r = tid + jj * 256;            // r = k*512 + i
    int k = r >> 9;
    int i = r & 511;
    float w = SCALE_CONV * (weight[(o * CIN + i) * KW + k] * s[b * CIN + i] + beta[b * CIN + i]);
    vals[jj] = w; sum += w; sq += w * w;
  }
  __shared__ float red1[256], red2[256];
  red1[tid] = sum; red2[tid] = sq;
  __syncthreads();
  for (int st = 128; st > 0; st >>= 1) {
    if (tid < st) { red1[tid] += red1[tid + st]; red2[tid] += red2[tid + st]; }
    __syncthreads();
  }
  float mean  = red1[0] * (1.0f / RED);
  float demod = rsqrtf(red2[0] + 1e-8f);
  __bf16* outp = wmod + (size_t)bo * RED;
#pragma unroll
  for (int jj = 0; jj < 6; ++jj) {
    int r = tid + jj * 256;
    outp[r] = (__bf16)((vals[jj] - mean) * demod);
  }
}

// ---------------- x: f32 [b][i][t]  ->  bf16 transposed padded [b][t+1][i] ----------------
__global__ void xpad_kernel(__bf16* __restrict__ xt) {
  int idx = blockIdx.x * 256 + threadIdx.x;                 // B_*2*CIN
  if (idx >= B_ * 2 * CIN) return;
  int b   = idx / (2 * CIN);
  int rem = idx - b * 2 * CIN;
  int half = rem / CIN;
  int i    = rem & (CIN - 1);
  size_t tp = half ? (size_t)(TP - 1) : 0;
  xt[((size_t)b * TP + tp) * CIN + i] = (__bf16)0.f;
}

__global__ void xcast_kernel(const float* __restrict__ x, __bf16* __restrict__ xt) {
  int idx = blockIdx.x * 256 + threadIdx.x;                 // B_*CIN*T_
  int b   = idx / (CIN * T_);
  int rem = idx - b * (CIN * T_);
  int i   = rem / T_;
  int t   = rem - i * T_;
  xt[((size_t)b * TP + (t + 1)) * CIN + i] = (__bf16)x[idx];
}

// ---------------- main GEMM (conv as GEMM) + fused GLU ----------------
// one wave computes a 16x64 output slab: 4 time tiles x (a-half + gate-half),
// A fragments reused across the 4 B fragments -> 1.5 b128 loads per WMMA
__global__ __launch_bounds__(256) void gemm_glu_kernel(const __bf16* __restrict__ wmod,
                                                       const __bf16* __restrict__ xt,
                                                       float* __restrict__ out) {
  int wave    = (blockIdx.x * 256 + (int)threadIdx.x) >> 5;
  int lane    = threadIdx.x & 31;
  int laneRow = lane & 15;
  int laneHi  = lane >> 4;
  int nt4 = wave & 31;                 // 2048/(16*NT) = 32 slabs in time
  int mt  = (wave >> 5) & 31;          // 512/16 GLU row tiles
  int b   = wave >> 10;                // batch
  int t0  = nt4 * (16 * NT);
  int m0  = mt * 16;

  // A lane layout (16-bit 16x32): row M = lane&15; K = {base..base+7, base+16..base+23}, base = laneHi*8
  const __bf16* Aa = wmod + ((size_t)(b * COUT2) + m0 + laneRow) * RED + laneHi * 8;
  const __bf16* Ag = Aa + (size_t)CGLU * RED;
  // B lane layout (16-bit 32x16): col N = lane&15; K = {base..base+7, base+16..base+23}, base = laneHi*8
  const __bf16* Bb = xt + ((size_t)b * TP + t0 + laneRow) * CIN + laneHi * 8;

  v8f acc_a[NT] = {};
  v8f acc_g[NT] = {};

  for (int k = 0; k < KW; ++k) {
    const __bf16* Aak = Aa + k * CIN;      // reduce offset k*512
    const __bf16* Agk = Ag + k * CIN;
    const __bf16* Bk  = Bb + k * CIN;      // time shift: padded col (+k rows of xt)
#pragma unroll
    for (int ic = 0; ic < 16; ++ic) {      // 16 channel chunks of 32
      Frag fa, fg;
      const __bf16* pa = Aak + ic * 32;
      fa.u[0] = *(const uint4v*)(pa);
      fa.u[1] = *(const uint4v*)(pa + 16);
      const __bf16* pg = Agk + ic * 32;
      fg.u[0] = *(const uint4v*)(pg);
      fg.u[1] = *(const uint4v*)(pg + 16);
#pragma unroll
      for (int j = 0; j < NT; ++j) {
        Frag fb;
        const __bf16* pb = Bk + (size_t)j * 16 * CIN + ic * 32;
        fb.u[0] = *(const uint4v*)(pb);
        fb.u[1] = *(const uint4v*)(pb + 16);
        acc_a[j] = __builtin_amdgcn_wmma_f32_16x16x32_bf16(false, fa.bf, false, fb.bf,
                                                           (short)0, acc_a[j], false, false);
        acc_g[j] = __builtin_amdgcn_wmma_f32_16x16x32_bf16(false, fg.bf, false, fb.bf,
                                                           (short)0, acc_g[j], false, false);
      }
    }
  }

  // C/D layout: VGPR r -> M = m0 + r + laneHi*8, N = t0 + j*16 + laneRow
  float* op = out + ((size_t)b * CGLU + m0 + laneHi * 8) * T_ + t0 + laneRow;
#pragma unroll
  for (int j = 0; j < NT; ++j) {
    float* opj = op + j * 16;
#pragma unroll
    for (int r = 0; r < 8; ++r) {
      float g   = acc_g[j][r];
      float sig = 1.0f / (1.0f + __expf(-g));
      opj[(size_t)r * T_] = acc_a[j][r] * sig;
    }
  }
}

extern "C" void kernel_launch(void* const* d_in, const int* in_sizes, int n_in,
                              void* d_out, int out_size, void* d_ws, size_t ws_size,
                              hipStream_t stream) {
  const float* x      = (const float*)d_in[0];
  // d_in[1] = c_src (unused by reference)
  const float* c_trg  = (const float*)d_in[2];
  const float* w_s    = (const float*)d_in[3];
  const float* b_s    = (const float*)d_in[4];
  const float* w_b    = (const float*)d_in[5];
  const float* b_b    = (const float*)d_in[6];
  const float* weight = (const float*)d_in[7];
  float* out = (float*)d_out;

  // workspace layout (all 256B aligned)
  float*  s_buf    = (float*)d_ws;                                   // 32 KB
  float*  beta_buf = s_buf + B_ * CIN;                               // 32 KB
  __bf16* wmod     = (__bf16*)((char*)d_ws + 65536);                 // 48 MB
  __bf16* xt       = (__bf16*)((char*)d_ws + 65536 +
                               (size_t)B_ * COUT2 * RED * 2);        // ~33.6 MB

  // 1) style linears
  style_kernel<<<(B_ * CIN + 255) / 256, 256, 0, stream>>>(
      c_trg, w_s, b_s, w_b, b_b, s_buf, beta_buf);

  // 2) modulated/demodulated weights -> bf16
  modw_kernel<<<B_ * COUT2, 256, 0, stream>>>(weight, s_buf, beta_buf, wmod);

  // 3) pad + transpose-cast x -> bf16 [b][t+1][i]
  xpad_kernel<<<(B_ * 2 * CIN + 255) / 256, 256, 0, stream>>>(xt);
  xcast_kernel<<<(B_ * CIN * T_) / 256, 256, 0, stream>>>(x, xt);

  // 4) WMMA GEMM + fused GLU: 16*32*32 = 16384 waves, 8 waves per block
  gemm_glu_kernel<<<(B_ * 32 * 32) / 8, 256, 0, stream>>>(wmod, xt, out);
}
